// SpatialExtentOutputHead_86337432584821
// MI455X (gfx1250) — matrix operand order repaired and checked by reading, not compiled, not run on hardware
//
#include <hip/hip_runtime.h>
#include <hip/hip_bf16.h>

typedef __attribute__((ext_vector_type(16))) __bf16 v16bf;
typedef __attribute__((ext_vector_type(8)))  float  v8f;

#define DIM 128

// ---------------- helpers ----------------

__device__ __forceinline__ unsigned short f2bfbits(float f) {
    __bf16 b = (__bf16)f;                       // native v_cvt (RNE)
    return __builtin_bit_cast(unsigned short, b);
}

// XOR-swizzled LDS address for bf16 W1^T: row j (128 elems), 16-elem chunks
__device__ __forceinline__ int swz(int j, int k) {
    return (j << 7) + ((((k >> 4) ^ (j & 7)) << 4) | (k & 15));
}

// ---------------- kernel 0: zero accumulators + output ----------------

__global__ void k_zero(float* accum, float* out, int bgraphs) {
    int idx = blockIdx.x * blockDim.x + threadIdx.x;
    if (idx < bgraphs * 4) accum[idx] = 0.f;
    if (idx < bgraphs)     out[idx]   = 0.f;
}

// ---------------- kernel 1: COM accumulation (wave-combined atomics) ------

__global__ void k_com_accum(const float* __restrict__ pos,
                            const float* __restrict__ masses_table,
                            const int*   __restrict__ an,
                            const int*   __restrict__ batch,
                            float* __restrict__ accum, int n) {
    int stride = gridDim.x * blockDim.x;
    int lane = threadIdx.x & 31;
    for (int i = blockIdx.x * blockDim.x + threadIdx.x; (i - lane) < n; i += stride) {
        bool active = i < n;
        int b = -1;
        float m = 0.f, vx = 0.f, vy = 0.f, vz = 0.f;
        if (active) {
            b = batch[i];
            m = masses_table[an[i]];
            vx = m * pos[3 * i + 0];
            vy = m * pos[3 * i + 1];
            vz = m * pos[3 * i + 2];
        }
        int b0 = __shfl(b, 0, 32);
        if (__all(b == b0)) {
            // whole wave in one segment: butterfly reduce, one atomic set
            #pragma unroll
            for (int off = 16; off > 0; off >>= 1) {
                vx += __shfl_xor(vx, off, 32);
                vy += __shfl_xor(vy, off, 32);
                vz += __shfl_xor(vz, off, 32);
                m  += __shfl_xor(m,  off, 32);
            }
            if (lane == 0) {
                atomicAdd(&accum[4 * b0 + 0], vx);
                atomicAdd(&accum[4 * b0 + 1], vy);
                atomicAdd(&accum[4 * b0 + 2], vz);
                atomicAdd(&accum[4 * b0 + 3], m);
            }
        } else if (active) {
            atomicAdd(&accum[4 * b + 0], vx);
            atomicAdd(&accum[4 * b + 1], vy);
            atomicAdd(&accum[4 * b + 2], vz);
            atomicAdd(&accum[4 * b + 3], m);
        }
    }
}

// ---------------- kernel 2: COM divide ----------------

__global__ void k_com_div(const float* __restrict__ accum,
                          float4* __restrict__ com, int bgraphs) {
    int b = blockIdx.x * blockDim.x + threadIdx.x;
    if (b < bgraphs) {
        float mt = accum[4 * b + 3];
        float inv = 1.f / mt;                    // empty graphs never gathered
        com[b] = make_float4(accum[4 * b + 0] * inv,
                             accum[4 * b + 1] * inv,
                             accum[4 * b + 2] * inv, 0.f);
    }
}

// ---------------- kernel 3: WMMA MLP + scatter reduce ----------------
// Steady state: all 32 B fragments (W1) + W2 weights live in VGPRs (~360 regs,
// uses the gfx1250 VGPR-MSB high-register space). Inner loop touches only the
// HBM energy stream + WMMA + epilogue math. LDS used once for transpose staging.

__global__ void __launch_bounds__(256)
k_main(const float* __restrict__ energy, const float* __restrict__ pos,
       const float* __restrict__ W1, const float* __restrict__ W2,
       const int* __restrict__ batch, const float4* __restrict__ com,
       float* __restrict__ out, int n) {
    __shared__ __align__(16) unsigned short sW1T[DIM * DIM]; // bf16, transposed, swizzled
    __shared__ float sW2[DIM];

    int t = threadIdx.x;
    // Stage W1^T (bf16) into LDS: idx = k*128 + j (coalesced global read)
    for (int idx = t; idx < DIM * DIM; idx += 256) {
        int k = idx >> 7, j = idx & 127;
        sW1T[swz(j, k)] = f2bfbits(W1[idx]);
    }
    if (t < DIM) sW2[t] = W2[t];
    __syncthreads();

    int lane = t & 31, wid = t >> 5;
    int half = lane >> 4;          // which 16-lane half
    int lq   = lane & 15;

    // Hoist all tile-invariant operands into registers.
    // B fragment (kc,jt): lanes 0-15 -> col j=jt*16+lq, K=kc*32+0..15
    //                     lanes16-31 -> col j=jt*16+lq, K=kc*32+16..31
    v16bf Bf[4][8];
    #pragma unroll
    for (int kc = 0; kc < 4; ++kc)
        #pragma unroll
        for (int jt = 0; jt < 8; ++jt)
            Bf[kc][jt] = *(const v16bf*)&sW1T[swz(jt * 16 + lq, kc * 32 + half * 16)];
    float w2r[8];
    #pragma unroll
    for (int jt = 0; jt < 8; ++jt) w2r[jt] = sW2[jt * 16 + lq];

    int ntiles = (n + 15) >> 4;
    int tstride = gridDim.x * 8;

    for (int tile = blockIdx.x * 8 + wid; tile < ntiles; tile += tstride) {
        int row = tile * 16 + lq;
        if (row >= n) row = n - 1;                    // clamp (masked at output)
        const float* rp = energy + (long long)row * DIM;

        // prefetch next tile's rows (wave-uniform guard)
        if (tile + tstride < ntiles) {
            int nrow = (tile + tstride) * 16 + lq;
            if (nrow >= n) nrow = n - 1;
            __builtin_prefetch(energy + (long long)nrow * DIM, 0, 1);
        }

        v8f acc[8] = {};
        #pragma unroll
        for (int kc = 0; kc < 4; ++kc) {
            // A fragment: lane<16 -> K {0..15}; lane>=16 -> K {16..31}
            const float4* p0 = (const float4*)(rp + kc * 32 + 16 * half);
            const float4* p1 = (const float4*)(rp + kc * 32 + 16 * half + 8);
            float4 a0 = p0[0], a1 = p0[1], c0 = p1[0], c1 = p1[1];
            v16bf A;
            A[0]  = (__bf16)a0.x; A[1]  = (__bf16)a0.y;
            A[2]  = (__bf16)a0.z; A[3]  = (__bf16)a0.w;
            A[4]  = (__bf16)a1.x; A[5]  = (__bf16)a1.y;
            A[6]  = (__bf16)a1.z; A[7]  = (__bf16)a1.w;
            A[8]  = (__bf16)c0.x; A[9]  = (__bf16)c0.y;
            A[10] = (__bf16)c0.z; A[11] = (__bf16)c0.w;
            A[12] = (__bf16)c1.x; A[13] = (__bf16)c1.y;
            A[14] = (__bf16)c1.z; A[15] = (__bf16)c1.w;

            #pragma unroll
            for (int jt = 0; jt < 8; ++jt) {
                acc[jt] = __builtin_amdgcn_wmma_f32_16x16x32_bf16(
                    false, A, false, Bf[kc][jt], (short)0, acc[jt], false, false);
            }
        }

        // epilogue: x_M = sum_j silu(h[M][j]) * W2[j]
        float s[8] = {};
        #pragma unroll
        for (int jt = 0; jt < 8; ++jt) {
            float w = w2r[jt];
            #pragma unroll
            for (int r = 0; r < 8; ++r) {
                float h = acc[jt][r];
                float sv = h * __builtin_amdgcn_rcpf(1.f + __expf(-h)); // silu, fast rcp
                s[r] += sv * w;
            }
        }
        // reduce over the 16 lanes of each half (C layout: lanes0-15 M=r, 16-31 M=r+8)
        #pragma unroll
        for (int r = 0; r < 8; ++r) {
            s[r] += __shfl_xor(s[r], 1, 32);
            s[r] += __shfl_xor(s[r], 2, 32);
            s[r] += __shfl_xor(s[r], 4, 32);
            s[r] += __shfl_xor(s[r], 8, 32);
        }

        if (lq == 0) {                 // lane 0 -> atoms M=0..7, lane 16 -> M=8..15
            int mbase = half * 8;
            int prevb = -1; float run = 0.f;
            #pragma unroll
            for (int r = 0; r < 8; ++r) {
                int atom = tile * 16 + mbase + r;
                if (atom < n) {
                    float px = pos[3 * atom + 0];
                    float py = pos[3 * atom + 1];
                    float pz = pos[3 * atom + 2];
                    int b = batch[atom];
                    float4 c = com[b];
                    float dx = px - c.x, dy = py - c.y, dz = pz - c.z;
                    float y = s[r] * (dx * dx + dy * dy + dz * dz);
                    if (b == prevb) run += y;
                    else { if (prevb >= 0) atomicAdd(&out[prevb], run); prevb = b; run = y; }
                }
            }
            if (prevb >= 0) atomicAdd(&out[prevb], run);
        }
    }
}

// ---------------- launcher ----------------

extern "C" void kernel_launch(void* const* d_in, const int* in_sizes, int n_in,
                              void* d_out, int out_size, void* d_ws, size_t ws_size,
                              hipStream_t stream) {
    const float* energy = (const float*)d_in[0];
    const float* pos    = (const float*)d_in[1];
    const float* W1     = (const float*)d_in[2];
    const float* W2     = (const float*)d_in[3];
    const float* masses = (const float*)d_in[4];
    const int*   an     = (const int*)d_in[5];
    const int*   batch  = (const int*)d_in[6];
    float* out = (float*)d_out;

    int n = in_sizes[0] / DIM;           // atoms
    int bgraphs = out_size;              // graphs

    float*  accum = (float*)d_ws;                                        // [B,4]
    float4* com   = (float4*)((char*)d_ws + (size_t)bgraphs * 4 * sizeof(float)); // [B]

    k_zero<<<(bgraphs * 4 + 255) / 256, 256, 0, stream>>>(accum, out, bgraphs);
    k_com_accum<<<2048, 256, 0, stream>>>(pos, masses, an, batch, accum, n);
    k_com_div<<<(bgraphs + 255) / 256, 256, 0, stream>>>(accum, com, bgraphs);
    k_main<<<1024, 256, 0, stream>>>(energy, pos, W1, W2, batch, com, out, n);
}
